// CausalSelfAttention_48043504173362
// MI455X (gfx1250) — compile-verified, matrix operand order checked by reading, not run on hardware
//
#include <hip/hip_runtime.h>
#include <hip/hip_bf16.h>

// Problem constants (reference: B=1, S=4096, E=1024, H=16, D=64)
constexpr int SQ = 4096;
constexpr int EE = 1024;
constexpr int HH = 16;
constexpr int DD = 64;

typedef __attribute__((ext_vector_type(16))) __bf16 v16bf;
typedef __attribute__((ext_vector_type(8)))  __bf16 v8bf;
typedef __attribute__((ext_vector_type(8)))  float  v8f;

union ABfrag {
    v16bf v;
    v8bf  h[2];
};

__device__ __forceinline__ unsigned short f2bf(float f) {
    unsigned int u = __builtin_bit_cast(unsigned int, f);
    u += 0x7FFFu + ((u >> 16) & 1u);       // round-to-nearest-even
    return (unsigned short)(u >> 16);
}

__device__ __forceinline__ v8f wmma_bf16(const ABfrag& a, const ABfrag& b, v8f c) {
    return __builtin_amdgcn_wmma_f32_16x16x32_bf16(
        /*neg_a=*/false, a.v, /*neg_b=*/false, b.v,
        /*c_mod=*/(short)0, c, /*reuse_a=*/false, /*reuse_b=*/false);
}

// CDNA5 async copy: global -> LDS, 16 bytes per lane, tracked by ASYNCcnt.
// lds_off = low 32 bits of the generic shared-memory address (LDS byte offset).
__device__ __forceinline__ void async_ld_b128(unsigned int lds_off,
                                              const unsigned short* gaddr) {
    asm volatile("global_load_async_to_lds_b128 %0, %1, off"
                 :: "v"(lds_off), "v"(gaddr) : "memory");
}

// ---------------------------------------------------------------------------
// fp32 -> bf16 cast
// ---------------------------------------------------------------------------
__global__ void cast_f32_to_bf16(const float* __restrict__ in,
                                 unsigned short* __restrict__ out, int n) {
    int i = blockIdx.x * blockDim.x + threadIdx.x;
    int stride = gridDim.x * blockDim.x;
    for (; i < n; i += stride) out[i] = f2bf(in[i]);
}

// ---------------------------------------------------------------------------
// bf16 WMMA GEMM: C[m,n] = sum_k A[m,k] * B[n,k]   (A: MxK, B: NxK row-major)
// Block = 4 waves = 64(M) x 64(N) tile. B panel (64x32) double-buffered in LDS
// via async global->LDS copies; each wave holds one A fragment (16xM rows) and
// 4 fp32 accumulators (4 N-tiles).
// MODE 0: store bf16 [H][M][64]  (n -> h = n>>6, d = n&63)   (Q, K)
// MODE 1: store bf16 [N][M]      (V transposed per head: [H][64][S])
// MODE 2: store fp32 [M][N]      (final output)
// ---------------------------------------------------------------------------
template <int MODE>
__global__ __launch_bounds__(128)
void gemm_bf16_wmma(const unsigned short* __restrict__ A,
                    const unsigned short* __restrict__ B,
                    void* __restrict__ C, int M, int N, int K) {
    __shared__ unsigned short Bsh[2][64 * 32];   // 2 x 4KB double buffer

    const int tid  = threadIdx.x;
    const int lane = tid & 31;
    const int wave = tid >> 5;
    const int n0   = blockIdx.x * 64;
    const int m0   = blockIdx.y * 64 + wave * 16;
    const int row  = lane & 15;
    const int kh   = (lane >> 4) * 8;  // fragment k-half offset (elements)

    const unsigned short* pa = A + (size_t)(m0 + row) * K;

    // Each thread stages two 16B chunks of the 64x32 B panel (256 chunks).
    const int r0 = tid >> 2,          s0e = (tid & 3) * 8;
    const int r1 = (tid + 128) >> 2,  s1e = ((tid + 128) & 3) * 8;

    v8f acc[4];
#pragma unroll
    for (int c = 0; c < 4; ++c) acc[c] = (v8f){0, 0, 0, 0, 0, 0, 0, 0};

    // prologue: stage k-step 0 into buffer 0
    async_ld_b128((unsigned int)(size_t)&Bsh[0][r0 * 32 + s0e],
                  B + (size_t)(n0 + r0) * K + s0e);
    async_ld_b128((unsigned int)(size_t)&Bsh[0][r1 * 32 + s1e],
                  B + (size_t)(n0 + r1) * K + s1e);

    const int nsteps = K / 32;
    for (int i = 0; i < nsteps; ++i) {
        const int k0 = i * 32;
        if (i + 1 < nsteps) {
            const int buf = (i + 1) & 1;
            async_ld_b128((unsigned int)(size_t)&Bsh[buf][r0 * 32 + s0e],
                          B + (size_t)(n0 + r0) * K + k0 + 32 + s0e);
            async_ld_b128((unsigned int)(size_t)&Bsh[buf][r1 * 32 + s1e],
                          B + (size_t)(n0 + r1) * K + k0 + 32 + s1e);
            asm volatile("s_wait_asynccnt 0x2" ::: "memory");  // current buf landed
        } else {
            asm volatile("s_wait_asynccnt 0x0" ::: "memory");
        }
        __syncthreads();

        ABfrag a;
        a.h[0] = *(const v8bf*)(pa + k0 + kh);
        a.h[1] = *(const v8bf*)(pa + k0 + kh + 16);

        const unsigned short* bb = Bsh[i & 1];
#pragma unroll
        for (int c = 0; c < 4; ++c) {
            ABfrag b;   // column n = n0 + c*16 + (lane&15)  <-> LDS row c*16+row
            b.h[0] = *(const v8bf*)&bb[(c * 16 + row) * 32 + kh];
            b.h[1] = *(const v8bf*)&bb[(c * 16 + row) * 32 + 16 + kh];
            acc[c] = wmma_bf16(a, b, acc[c]);
        }
        __syncthreads();   // compute done before this buffer is overwritten
    }

    // C layout: VGPR r -> M = r (lanes 0-15) / r+8 (lanes 16-31), N = lane&15
    const int cmb = m0 + (lane >> 4) * 8;
#pragma unroll
    for (int c = 0; c < 4; ++c) {
        const int cn = n0 + c * 16 + (lane & 15);
#pragma unroll
        for (int r = 0; r < 8; ++r) {
            const int cm = cmb + r;
            const float val = acc[c][r];
            if (MODE == 0) {
                ((unsigned short*)C)[((size_t)(cn >> 6) * M + cm) * 64 + (cn & 63)] = f2bf(val);
            } else if (MODE == 1) {
                ((unsigned short*)C)[(size_t)cn * M + cm] = f2bf(val);
            } else {
                ((float*)C)[(size_t)cm * N + cn] = val;
            }
        }
    }
}

// ---------------------------------------------------------------------------
// Flash attention (causal, online softmax), bf16 WMMA.
// Q,K: [H][S][64] bf16.  Vt: [H][64][S] bf16.  AO out: [S][E] bf16.
// Block = 4 waves; wave w handles query rows [ (bx*4+w)*16, +16 ) of head by.
// ---------------------------------------------------------------------------
__global__ __launch_bounds__(128)
void attn_wmma(const unsigned short* __restrict__ Q,
               const unsigned short* __restrict__ Km,
               const unsigned short* __restrict__ Vt,
               unsigned short* __restrict__ AO) {
    __shared__ unsigned short Pl[4][16 * 32];

    const int lane = threadIdx.x & 31;
    const int wave = threadIdx.x >> 5;
    const int h    = blockIdx.y;
    const int q0   = (blockIdx.x * 4 + wave) * 16;
    const int col  = lane & 15;
    const int hb   = lane >> 4;
    const int kh   = hb * 8;

    const unsigned short* Qh = Q  + (size_t)h * SQ * 64;
    const unsigned short* Kh = Km + (size_t)h * SQ * 64;
    const unsigned short* Vh = Vt + (size_t)h * 64 * SQ;

    // Q fragments for the whole tile (D=64 -> two k=32 fragments)
    const unsigned short* qrow = Qh + (size_t)(q0 + col) * 64;
    ABfrag qf0, qf1;
    qf0.h[0] = *(const v8bf*)(qrow + kh);
    qf0.h[1] = *(const v8bf*)(qrow + kh + 16);
    qf1.h[0] = *(const v8bf*)(qrow + 32 + kh);
    qf1.h[1] = *(const v8bf*)(qrow + 32 + kh + 16);

    float rmax[8], rsum[8];
    v8f accO[4];
#pragma unroll
    for (int r = 0; r < 8; ++r) { rmax[r] = -3.0e38f; rsum[r] = 0.0f; }
#pragma unroll
    for (int c = 0; c < 4; ++c) accO[c] = (v8f){0, 0, 0, 0, 0, 0, 0, 0};

    const float scale = 0.125f;  // 1/sqrt(64)
    const int kmax = q0 + 16;

    for (int kb = 0; kb < kmax; kb += 32) {
        // ---- scores: S[16x32] = Q(16x64) x K^T, two 16x16 tiles ----
        const unsigned short* k0r = Kh + (size_t)(kb + col) * 64;
        const unsigned short* k1r = Kh + (size_t)(kb + 16 + col) * 64;
        if (kb + 32 < kmax)
            __builtin_prefetch(Kh + (size_t)(kb + 32 + col) * 64, 0, 0);

        v8f s0 = (v8f){0, 0, 0, 0, 0, 0, 0, 0};
        v8f s1 = (v8f){0, 0, 0, 0, 0, 0, 0, 0};
        ABfrag bf;
        bf.h[0] = *(const v8bf*)(k0r + kh);
        bf.h[1] = *(const v8bf*)(k0r + kh + 16);
        s0 = wmma_bf16(qf0, bf, s0);
        bf.h[0] = *(const v8bf*)(k0r + 32 + kh);
        bf.h[1] = *(const v8bf*)(k0r + 32 + kh + 16);
        s0 = wmma_bf16(qf1, bf, s0);
        bf.h[0] = *(const v8bf*)(k1r + kh);
        bf.h[1] = *(const v8bf*)(k1r + kh + 16);
        s1 = wmma_bf16(qf0, bf, s1);
        bf.h[0] = *(const v8bf*)(k1r + 32 + kh);
        bf.h[1] = *(const v8bf*)(k1r + 32 + kh + 16);
        s1 = wmma_bf16(qf1, bf, s1);

        // ---- online softmax (row = r + hb*8, spread across 16-lane half) ----
        float p0[8], p1[8];
#pragma unroll
        for (int r = 0; r < 8; ++r) {
            const int qr = q0 + r + hb * 8;
            float a0 = s0[r] * scale;
            float a1 = s1[r] * scale;
            if (kb + col      > qr) a0 = -3.0e38f;   // causal mask
            if (kb + 16 + col > qr) a1 = -3.0e38f;
            float t = fmaxf(a0, a1);
#pragma unroll
            for (int m = 8; m >= 1; m >>= 1) t = fmaxf(t, __shfl_xor(t, m, 16));
            const float mnew = fmaxf(rmax[r], t);
            const float fac  = __expf(rmax[r] - mnew);
            rmax[r] = mnew;
            p0[r] = __expf(a0 - mnew);
            p1[r] = __expf(a1 - mnew);
            float ps = p0[r] + p1[r];
#pragma unroll
            for (int m = 8; m >= 1; m >>= 1) ps += __shfl_xor(ps, m, 16);
            rsum[r] = rsum[r] * fac + ps;
#pragma unroll
            for (int c = 0; c < 4; ++c) accO[c][r] *= fac;
        }

        // ---- P: C-layout -> A-fragment transpose through LDS ----
        unsigned short* pl = Pl[wave];
#pragma unroll
        for (int r = 0; r < 8; ++r) {
            const int m = r + hb * 8;
            pl[m * 32 + col]      = f2bf(p0[r]);
            pl[m * 32 + 16 + col] = f2bf(p1[r]);
        }
        ABfrag pf;  // lane row = col, k-slots {kh..kh+7, 16+kh..16+kh+7}
        pf.h[0] = *(const v8bf*)&pl[col * 32 + kh];
        pf.h[1] = *(const v8bf*)&pl[col * 32 + 16 + kh];

        // ---- O += P(16x32) x V(32x64): 4 N-tiles of 16, V pre-transposed ----
#pragma unroll
        for (int c = 0; c < 4; ++c) {
            const unsigned short* vr = Vh + (size_t)(c * 16 + col) * SQ + kb;
            ABfrag vf;
            vf.h[0] = *(const v8bf*)(vr + kh);
            vf.h[1] = *(const v8bf*)(vr + 16 + kh);
            accO[c] = wmma_bf16(pf, vf, accO[c]);
        }
    }

    // ---- normalize and store bf16 row-major [S][E] ----
#pragma unroll
    for (int r = 0; r < 8; ++r) {
        const int m = q0 + r + hb * 8;
        const float inv = 1.0f / rsum[r];
#pragma unroll
        for (int c = 0; c < 4; ++c)
            AO[(size_t)m * EE + h * 64 + c * 16 + col] = f2bf(accO[c][r] * inv);
    }
}

// ---------------------------------------------------------------------------
extern "C" void kernel_launch(void* const* d_in, const int* in_sizes, int n_in,
                              void* d_out, int out_size, void* d_ws, size_t ws_size,
                              hipStream_t stream) {
    const float* x  = (const float*)d_in[0];
    const float* Wq = (const float*)d_in[1];
    const float* Wk = (const float*)d_in[2];
    const float* Wv = (const float*)d_in[3];
    const float* Wo = (const float*)d_in[4];

    // Workspace layout (bf16 as ushort): 48 MB total
    unsigned short* xb  = (unsigned short*)d_ws;          // S*E
    unsigned short* wqb = xb  + (size_t)SQ * EE;          // E*E
    unsigned short* wkb = wqb + (size_t)EE * EE;
    unsigned short* wvb = wkb + (size_t)EE * EE;
    unsigned short* wob = wvb + (size_t)EE * EE;
    unsigned short* Qb  = wob + (size_t)EE * EE;          // [H][S][64]
    unsigned short* Kb  = Qb  + (size_t)SQ * EE;          // [H][S][64]
    unsigned short* Vtb = Kb  + (size_t)SQ * EE;          // [H][64][S]
    unsigned short* AOb = Vtb + (size_t)SQ * EE;          // [S][E]

    cast_f32_to_bf16<<<2048, 256, 0, stream>>>(x,  xb,  SQ * EE);
    cast_f32_to_bf16<<<1024, 256, 0, stream>>>(Wq, wqb, EE * EE);
    cast_f32_to_bf16<<<1024, 256, 0, stream>>>(Wk, wkb, EE * EE);
    cast_f32_to_bf16<<<1024, 256, 0, stream>>>(Wv, wvb, EE * EE);
    cast_f32_to_bf16<<<1024, 256, 0, stream>>>(Wo, wob, EE * EE);

    dim3 gblk(128);
    dim3 ggrd(EE / 64, SQ / 64);
    gemm_bf16_wmma<0><<<ggrd, gblk, 0, stream>>>(xb, wqb, Qb,  SQ, EE, EE);
    gemm_bf16_wmma<0><<<ggrd, gblk, 0, stream>>>(xb, wkb, Kb,  SQ, EE, EE);
    gemm_bf16_wmma<1><<<ggrd, gblk, 0, stream>>>(xb, wvb, Vtb, SQ, EE, EE);

    attn_wmma<<<dim3(SQ / 64, HH), 128, 0, stream>>>(Qb, Kb, Vtb, AOb);

    gemm_bf16_wmma<2><<<ggrd, gblk, 0, stream>>>(AOb, wob, d_out, SQ, EE, EE);
}
